// ConformerBlock_47459388621652
// MI455X (gfx1250) — matrix operand order repaired and checked by reading, not compiled, not run on hardware
//
#include <hip/hip_runtime.h>
#include <cstdint>
#include <cstddef>

#define B_   8
#define T_   1024
#define D_   512
#define H_   8
#define DH_  64
#define DFF_ 2048
#define KW_  31
#define NTOK (B_*T_)

typedef __attribute__((ext_vector_type(16))) __bf16 v16bf;
typedef __attribute__((ext_vector_type(8)))  float  v8f;
typedef unsigned short u16;
typedef unsigned int   u32;

// ---------- bf16 helpers (RNE convert, bit reinterpret) ----------
__device__ __forceinline__ u16 f2bf(float f){
    u32 u = __float_as_uint(f);
    u32 r = u + 0x7FFFu + ((u >> 16) & 1u);
    return (u16)(r >> 16);
}
__device__ __forceinline__ __bf16 bfbits(u16 s){
    union { u16 u; __bf16 b; } c; c.u = s; return c.b;
}
// CDNA5 WMMA bf16 16x16x32 fragment index maps (ISA 7.12.2):
// A (16x32, 16-bit): lanes0-15 hold K0-7/K16-23, lanes16-31 K8-15/K24-31
__device__ __forceinline__ int kidxA(int e, int half){ return ((e >> 3) << 4) + (half << 3) + (e & 7); }
// B (32x16, 16-bit): lanes0-15 hold K0-7/K16-23, lanes16-31 K8-15/K24-31 (row striped across lanes)
__device__ __forceinline__ int kidxB(int e, int half){ return ((e >> 3) << 4) + (half << 3) + (e & 7); }
// C/D (16x16 f32): element e -> row = e + 8*half, col = lane&15

__device__ __forceinline__ v8f wmma_bf16(v16bf a, v16bf b, v8f c){
    return __builtin_amdgcn_wmma_f32_16x16x32_bf16(false, a, false, b, (short)0, c, false, false);
}

// =================== LayerNorm (rowwise over D=512) ===================
// MODE 0: write bf16; MODE 1: write bf16 + sinusoidal PE; MODE 2: write f32
template<int MODE>
__global__ __launch_bounds__(256) void ln_kernel(const float* __restrict__ x,
                                                 const float* __restrict__ g,
                                                 const float* __restrict__ b,
                                                 u16* __restrict__ obf,
                                                 float* __restrict__ of){
    __shared__ float red[256];
    const int row = blockIdx.x;
    const int tid = threadIdx.x;
    const float* xr = x + (size_t)row * D_;
    float v0 = xr[tid], v1 = xr[tid + 256];
    red[tid] = v0 + v1; __syncthreads();
    for (int st = 128; st > 0; st >>= 1){ if (tid < st) red[tid] += red[tid + st]; __syncthreads(); }
    float mean = red[0] * (1.f / D_);
    __syncthreads();
    float d0 = v0 - mean, d1 = v1 - mean;
    red[tid] = d0 * d0 + d1 * d1; __syncthreads();
    for (int st = 128; st > 0; st >>= 1){ if (tid < st) red[tid] += red[tid + st]; __syncthreads(); }
    float rs = rsqrtf(red[0] * (1.f / D_) + 1e-5f);
    const int t = row & (T_ - 1);
    #pragma unroll
    for (int p = 0; p < 2; ++p){
        int dd = tid + p * 256;
        float v = (p == 0) ? v0 : v1;
        float y = (v - mean) * rs * g[dd] + b[dd];
        if (MODE == 1){
            int i2 = dd >> 1;
            float ang = (float)t * __powf(10000.f, -(float)(2 * i2) / (float)D_);
            y += (dd & 1) ? __cosf(ang) : __sinf(ang);
        }
        if (MODE == 2) of[(size_t)row * D_ + dd] = y;
        else           obf[(size_t)row * D_ + dd] = f2bf(y);
    }
}

// =================== Generic WMMA GEMM: C = A(bf16)[M,K] x W(f32) + epilogue ===================
// BL: 0 = W row-major [K,N]; 1 = head-major [H,K,64] (n=(h,e)); 2 = transposed [N,K]
// EPI: 0 outF=acc+bias; 1 outB=bf16(silu(acc+bias)); 2 outF=1.5*res+0.5*(acc+bias);
//      3 outF=2*res+acc+bias; 4 outB bf16 head-scatter to [B,H,T,64]
template<int BL, int EPI>
__global__ __launch_bounds__(256) void gemm_kernel(const u16*   __restrict__ A,
                                                   const float* __restrict__ Bw,
                                                   const float* __restrict__ bias,
                                                   const float* __restrict__ res,
                                                   float* __restrict__ outF,
                                                   u16*   __restrict__ outB,
                                                   int M, int N, int K){
    __shared__ u16 As[128][40];   // 128(M) x 32(K), padded
    __shared__ u16 Bs[32][72];    // 32(K) x 64(N), padded
    const int m0 = blockIdx.y * 128;
    const int n0 = blockIdx.x * 64;
    const int tid  = threadIdx.x;
    const int lane = tid & 31, wave = tid >> 5;
    const int half = lane >> 4, lr = lane & 15;
    const int wm = wave >> 1, wn = wave & 1;          // 4x2 wave grid -> each wave 32x32
    const int arow = tid >> 1, ajb = (tid & 1) * 16;  // A tile load map
    const int brow = tid >> 3, bcb = (tid & 7) * 8;   // B tile load map

    v8f acc[2][2] = {};

    for (int kt = 0; kt < K; kt += 32){
        __syncthreads();
        const u16* Ag = A + (size_t)(m0 + arow) * K + kt + ajb;
        #pragma unroll
        for (int j = 0; j < 16; ++j) As[arow][ajb + j] = Ag[j];
        #pragma unroll
        for (int c = 0; c < 8; ++c){
            int n = n0 + bcb + c;
            int k = kt + brow;
            float w;
            if (BL == 0)      w = Bw[(size_t)k * N + n];
            else if (BL == 1) w = Bw[((size_t)(n >> 6) * K + k) * 64 + (n & 63)];
            else              w = Bw[(size_t)n * K + k];
            Bs[brow][bcb + c] = f2bf(w);
        }
        __syncthreads();
        v16bf af[2], bfv[2];
        #pragma unroll
        for (int f = 0; f < 2; ++f){
            #pragma unroll
            for (int e = 0; e < 16; ++e){
                af[f][e]  = bfbits(As[wm * 32 + f * 16 + lr][kidxA(e, half)]);
                bfv[f][e] = bfbits(Bs[kidxB(e, half)][wn * 32 + f * 16 + lr]);
            }
        }
        #pragma unroll
        for (int fm = 0; fm < 2; ++fm)
            #pragma unroll
            for (int fn = 0; fn < 2; ++fn)
                acc[fm][fn] = wmma_bf16(af[fm], bfv[fn], acc[fm][fn]);
    }

    #pragma unroll
    for (int fm = 0; fm < 2; ++fm)
    #pragma unroll
    for (int fn = 0; fn < 2; ++fn)
    #pragma unroll
    for (int e = 0; e < 8; ++e){
        int gm = m0 + wm * 32 + fm * 16 + half * 8 + e;
        int gn = n0 + wn * 32 + fn * 16 + lr;
        float v = acc[fm][fn][e] + (bias ? bias[gn] : 0.f);
        size_t oi = (size_t)gm * N + gn;
        if (EPI == 0)      outF[oi] = v;
        else if (EPI == 1){ float s = v / (1.f + __expf(-v)); outB[oi] = f2bf(s); }
        else if (EPI == 2) outF[oi] = 1.5f * res[oi] + 0.5f * v;
        else if (EPI == 3) outF[oi] = 2.f * res[oi] + v;
        else {
            int bb = gm >> 10, t = gm & (T_ - 1);
            int h  = gn >> 6,  ee = gn & 63;
            outB[(((size_t)bb * H_ + h) * T_ + t) * DH_ + ee] = f2bf(v);
        }
    }
}

// =================== Flash attention: per (b,h), 64 query rows / block ===================
__global__ __launch_bounds__(128) void attn_kernel(const u16* __restrict__ Q,
                                                   const u16* __restrict__ Kb,
                                                   const u16* __restrict__ Vb,
                                                   u16* __restrict__ O){
    __shared__ u16 Ks[64][72];
    __shared__ u16 Vs[64][72];
    __shared__ u16 Ps[4][16][72];
    const int bh   = blockIdx.y;
    const int tid  = threadIdx.x;
    const int wave = tid >> 5, lane = tid & 31;
    const int half = lane >> 4, lr = lane & 15;
    const int q0   = blockIdx.x * 64 + wave * 16;
    const size_t base = (size_t)bh * T_ * DH_;
    const float scale = 0.125f;  // 1/sqrt(64)

    v16bf qf[2];
    #pragma unroll
    for (int kc = 0; kc < 2; ++kc)
        #pragma unroll
        for (int e = 0; e < 16; ++e)
            qf[kc][e] = bfbits(Q[base + (size_t)(q0 + lr) * DH_ + kc * 32 + kidxA(e, half)]);

    float mrun[8], lrun[8];
    v8f oacc[4] = {};
    #pragma unroll
    for (int e = 0; e < 8; ++e){ mrun[e] = -3.0e38f; lrun[e] = 0.f; }

    for (int kt = 0; kt < T_ / 64; ++kt){
        __syncthreads();
        for (int i = tid; i < 64 * 64; i += 128){
            int r = i >> 6, c = i & 63;
            Ks[r][c] = Kb[base + (size_t)(kt * 64 + r) * DH_ + c];
            Vs[r][c] = Vb[base + (size_t)(kt * 64 + r) * DH_ + c];
        }
        __syncthreads();

        v8f s[4];
        #pragma unroll
        for (int nt = 0; nt < 4; ++nt){
            v8f c8 = {};
            #pragma unroll
            for (int kc = 0; kc < 2; ++kc){
                v16bf kf;
                #pragma unroll
                for (int e = 0; e < 16; ++e)
                    kf[e] = bfbits(Ks[nt * 16 + lr][kc * 32 + kidxB(e, half)]);
                c8 = wmma_bf16(qf[kc], kf, c8);
            }
            s[nt] = c8;
        }
        // online softmax; row of element e is q0 + 8*half + e, cols across 16 lanes of the half
        #pragma unroll
        for (int e = 0; e < 8; ++e){
            float mx = -3.0e38f;
            #pragma unroll
            for (int nt = 0; nt < 4; ++nt){ float sv = s[nt][e] * scale; s[nt][e] = sv; mx = fmaxf(mx, sv); }
            #pragma unroll
            for (int msk = 1; msk < 16; msk <<= 1) mx = fmaxf(mx, __shfl_xor(mx, msk));
            float mn   = fmaxf(mrun[e], mx);
            float corr = __expf(mrun[e] - mn);
            float rsum = 0.f;
            #pragma unroll
            for (int nt = 0; nt < 4; ++nt){
                float p = __expf(s[nt][e] - mn);
                rsum += p;
                Ps[wave][half * 8 + e][nt * 16 + lr] = f2bf(p);
            }
            #pragma unroll
            for (int msk = 1; msk < 16; msk <<= 1) rsum += __shfl_xor(rsum, msk);
            lrun[e] = lrun[e] * corr + rsum;
            mrun[e] = mn;
            #pragma unroll
            for (int dt = 0; dt < 4; ++dt) oacc[dt][e] = oacc[dt][e] * corr;
        }
        __syncthreads();
        // O += P x V
        #pragma unroll
        for (int dt = 0; dt < 4; ++dt){
            #pragma unroll
            for (int kc = 0; kc < 2; ++kc){
                v16bf pa, vv;
                #pragma unroll
                for (int e = 0; e < 16; ++e){
                    pa[e] = bfbits(Ps[wave][lr][kc * 32 + kidxA(e, half)]);
                    vv[e] = bfbits(Vs[kc * 32 + kidxB(e, half)][dt * 16 + lr]);
                }
                oacc[dt] = wmma_bf16(pa, vv, oacc[dt]);
            }
        }
    }
    #pragma unroll
    for (int dt = 0; dt < 4; ++dt)
        #pragma unroll
        for (int e = 0; e < 8; ++e){
            float v = oacc[dt][e] / lrun[e];
            O[base + (size_t)(q0 + half * 8 + e) * DH_ + dt * 16 + lr] = f2bf(v);
        }
}

// =================== Conv module helpers ===================
// GLU: y1 [8192, 1024] token-major f32 -> channel-major bf16 [B, 512, T]
__global__ __launch_bounds__(256) void glu_kernel(const float* __restrict__ Y1, u16* __restrict__ Xcm){
    int idx = blockIdx.x * 256 + threadIdx.x;
    if (idx >= B_ * T_ * D_) return;
    int d = idx % D_, bt = idx / D_;
    int t = bt % T_,  bb = bt / T_;
    float a = Y1[(size_t)bt * (2 * D_) + d];
    float g = Y1[(size_t)bt * (2 * D_) + D_ + d];
    float v = a / (1.f + __expf(-g));
    Xcm[((size_t)bb * D_ + d) * T_ + t] = f2bf(v);
}

// 31-tap dense conv over time as WMMA GEMM: Y[b,e,t] = sum_k sum_d W[e,d,k]*X[b,d,t+k-15]
__global__ __launch_bounds__(256) void conv_kernel(const float* __restrict__ Wd,
                                                   const u16*   __restrict__ Xcm,
                                                   float* __restrict__ Ycm){
    __shared__ u16 As[128][40];
    __shared__ u16 Bs[32][72];
    const int n0 = blockIdx.x * 64;    // time tile
    const int m0 = blockIdx.y * 128;   // out-channel tile
    const int bb = blockIdx.z;
    const int tid  = threadIdx.x;
    const int lane = tid & 31, wave = tid >> 5;
    const int half = lane >> 4, lr = lane & 15;
    const int wm = wave >> 1, wn = wave & 1;
    const int arow = tid >> 1, ajb = (tid & 1) * 16;
    const int brow = tid >> 3, bcb = (tid & 7) * 8;

    v8f acc[2][2] = {};

    for (int tap = 0; tap < KW_; ++tap){
        const int tshift = tap - KW_ / 2;
        for (int dc = 0; dc < D_; dc += 32){
            __syncthreads();
            #pragma unroll
            for (int j = 0; j < 16; ++j){
                int d = dc + ajb + j;
                As[arow][ajb + j] = f2bf(Wd[((size_t)(m0 + arow) * D_ + d) * KW_ + tap]);
            }
            #pragma unroll
            for (int c = 0; c < 8; ++c){
                int t = n0 + bcb + c + tshift;
                int d = dc + brow;
                u16 val = 0;
                if (t >= 0 && t < T_) val = Xcm[((size_t)bb * D_ + d) * T_ + t];
                Bs[brow][bcb + c] = val;
            }
            __syncthreads();
            v16bf af[2], bfv[2];
            #pragma unroll
            for (int f = 0; f < 2; ++f){
                #pragma unroll
                for (int e = 0; e < 16; ++e){
                    af[f][e]  = bfbits(As[wm * 32 + f * 16 + lr][kidxA(e, half)]);
                    bfv[f][e] = bfbits(Bs[kidxB(e, half)][wn * 32 + f * 16 + lr]);
                }
            }
            #pragma unroll
            for (int fm = 0; fm < 2; ++fm)
                #pragma unroll
                for (int fn = 0; fn < 2; ++fn)
                    acc[fm][fn] = wmma_bf16(af[fm], bfv[fn], acc[fm][fn]);
        }
    }
    #pragma unroll
    for (int fm = 0; fm < 2; ++fm)
    #pragma unroll
    for (int fn = 0; fn < 2; ++fn)
    #pragma unroll
    for (int e = 0; e < 8; ++e){
        int gm = m0 + wm * 32 + fm * 16 + half * 8 + e;
        int gn = n0 + wn * 32 + fn * 16 + lr;
        Ycm[((size_t)bb * D_ + gm) * T_ + gn] = acc[fm][fn][e];
    }
}

// BatchNorm stats over (batch,time) per channel, biased variance
__global__ __launch_bounds__(256) void bn_stats_kernel(const float* __restrict__ Y, float* __restrict__ st){
    __shared__ float r1[256], r2[256];
    const int ch = blockIdx.x, tid = threadIdx.x;
    float s = 0.f, s2 = 0.f;
    for (int i = tid; i < B_ * T_; i += 256){
        int bb = i >> 10, t = i & (T_ - 1);
        float v = Y[((size_t)bb * D_ + ch) * T_ + t];
        s += v; s2 += v * v;
    }
    r1[tid] = s; r2[tid] = s2; __syncthreads();
    for (int stp = 128; stp > 0; stp >>= 1){
        if (tid < stp){ r1[tid] += r1[tid + stp]; r2[tid] += r2[tid + stp]; }
        __syncthreads();
    }
    if (tid == 0){
        float m = r1[0] / (float)(B_ * T_);
        st[ch] = m;
        st[D_ + ch] = r2[0] / (float)(B_ * T_) - m * m;
    }
}

// BN apply + SiLU, transpose channel-major f32 -> token-major bf16
__global__ __launch_bounds__(256) void bn_silu_kernel(const float* __restrict__ Y,
                                                      const float* __restrict__ st,
                                                      const float* __restrict__ g,
                                                      const float* __restrict__ b,
                                                      u16* __restrict__ Otok){
    int idx = blockIdx.x * 256 + threadIdx.x;
    if (idx >= B_ * D_ * T_) return;
    int t = idx % T_, rem = idx / T_;
    int ch = rem % D_, bb = rem / D_;
    float v = Y[idx];
    float y = (v - st[ch]) * rsqrtf(st[D_ + ch] + 1e-5f) * g[ch] + b[ch];
    float s = y / (1.f + __expf(-y));
    Otok[((size_t)bb * T_ + t) * D_ + ch] = f2bf(s);
}

// =================== Host orchestration ===================
extern "C" void kernel_launch(void* const* d_in, const int* in_sizes, int n_in,
                              void* d_out, int out_size, void* d_ws, size_t ws_size,
                              hipStream_t stream){
    (void)in_sizes; (void)n_in; (void)out_size; (void)ws_size;
    const float* x      = (const float*)d_in[0];
    const float* ff1_g  = (const float*)d_in[1];
    const float* ff1_b  = (const float*)d_in[2];
    const float* ff1_w1 = (const float*)d_in[3];
    const float* ff1_b1 = (const float*)d_in[4];
    const float* ff1_w2 = (const float*)d_in[5];
    const float* ff1_b2 = (const float*)d_in[6];
    const float* mha_g  = (const float*)d_in[7];
    const float* mha_b  = (const float*)d_in[8];
    const float* wq     = (const float*)d_in[9];
    const float* bq     = (const float*)d_in[10];
    const float* wk     = (const float*)d_in[11];
    const float* bk     = (const float*)d_in[12];
    const float* wv     = (const float*)d_in[13];
    const float* bv     = (const float*)d_in[14];
    const float* wo     = (const float*)d_in[15];
    const float* bo     = (const float*)d_in[16];
    const float* cv_g   = (const float*)d_in[17];
    const float* cv_b   = (const float*)d_in[18];
    const float* cv_w1  = (const float*)d_in[19];
    const float* cv_b1  = (const float*)d_in[20];
    const float* cv_wd  = (const float*)d_in[21];
    const float* bn_g   = (const float*)d_in[22];
    const float* bn_b   = (const float*)d_in[23];
    const float* cv_w2  = (const float*)d_in[24];
    const float* cv_b2  = (const float*)d_in[25];
    const float* ff2_g  = (const float*)d_in[26];
    const float* ff2_b  = (const float*)d_in[27];
    const float* ff2_w1 = (const float*)d_in[28];
    const float* ff2_b1 = (const float*)d_in[29];
    const float* ff2_w2 = (const float*)d_in[30];
    const float* ff2_b2 = (const float*)d_in[31];
    const float* fin_g  = (const float*)d_in[32];
    const float* fin_b  = (const float*)d_in[33];

    char* W = (char*)d_ws;
    size_t off = 0;
    auto take = [&](size_t bytes) -> char* {
        char* p = W + off;
        off += (bytes + 255) & ~(size_t)255;
        return p;
    };
    float* x_res = (float*)take((size_t)NTOK * D_ * 4);   // residual stream (f32)
    u16*   lnA   = (u16*)  take((size_t)NTOK * D_ * 2);   // LN output (bf16 GEMM A)
    char*  S     =         take((size_t)68 << 20);        // per-module scratch (overlaid)

    // overlays within S (sequential pipeline stages)
    u16*   hbf    = (u16*)S;                                                 // FF hidden bf16 [8192,2048]
    u16*   qb     = (u16*)S;
    u16*   kb     = (u16*)(S + 1 * ((size_t)NTOK * D_ * 2));
    u16*   vb     = (u16*)(S + 2 * ((size_t)NTOK * D_ * 2));
    u16*   ob     = (u16*)(S + 3 * ((size_t)NTOK * D_ * 2));
    float* y1     = (float*)S;                                               // pw1 out f32 [8192,1024]
    u16*   glu_cm = (u16*)(S + (size_t)NTOK * 2 * D_ * 4);                   // bf16 [B,512,T]
    float* conv_o = (float*)(S + (size_t)NTOK * 2 * D_ * 4 + (size_t)NTOK * D_ * 2);
    float* bn_st  = (float*)((char*)conv_o + (size_t)NTOK * D_ * 4);
    u16*   bn_act = (u16*)((char*)bn_st + 4096);

    const dim3 g512(D_ / 64, NTOK / 128);       // N=512 GEMMs
    const dim3 gDFF(DFF_ / 64, NTOK / 128);     // N=2048 GEMMs
    const dim3 g2D(2 * D_ / 64, NTOK / 128);    // N=1024 GEMM

    // ---- FF module 1: x = 1.5*x + 0.5*(silu(ln(x)W1+b1)W2+b2) ----
    ln_kernel<0><<<NTOK, 256, 0, stream>>>(x, ff1_g, ff1_b, lnA, nullptr);
    gemm_kernel<0, 1><<<gDFF, 256, 0, stream>>>(lnA, ff1_w1, ff1_b1, nullptr, nullptr, hbf, NTOK, DFF_, D_);
    gemm_kernel<0, 2><<<g512, 256, 0, stream>>>(hbf, ff1_w2, ff1_b2, x, x_res, nullptr, NTOK, D_, DFF_);

    // ---- MHSA: y = ln(x)+PE; q/k/v; flash attention; x = 2x + (oW_o + b_o) ----
    ln_kernel<1><<<NTOK, 256, 0, stream>>>(x_res, mha_g, mha_b, lnA, nullptr);
    gemm_kernel<1, 4><<<g512, 256, 0, stream>>>(lnA, wq, bq, nullptr, nullptr, qb, NTOK, D_, D_);
    gemm_kernel<1, 4><<<g512, 256, 0, stream>>>(lnA, wk, bk, nullptr, nullptr, kb, NTOK, D_, D_);
    gemm_kernel<1, 4><<<g512, 256, 0, stream>>>(lnA, wv, bv, nullptr, nullptr, vb, NTOK, D_, D_);
    attn_kernel<<<dim3(T_ / 64, B_ * H_), 128, 0, stream>>>(qb, kb, vb, ob);
    gemm_kernel<0, 3><<<g512, 256, 0, stream>>>(ob, wo, bo, x_res, x_res, nullptr, NTOK, D_, D_);

    // ---- Conv module: pw1 -> GLU -> conv(K=31) -> BN -> SiLU -> pw2; x = 2x + y ----
    ln_kernel<0><<<NTOK, 256, 0, stream>>>(x_res, cv_g, cv_b, lnA, nullptr);
    gemm_kernel<2, 0><<<g2D, 256, 0, stream>>>(lnA, cv_w1, cv_b1, nullptr, y1, nullptr, NTOK, 2 * D_, D_);
    glu_kernel<<<(B_ * T_ * D_ + 255) / 256, 256, 0, stream>>>(y1, glu_cm);
    conv_kernel<<<dim3(T_ / 64, D_ / 128, B_), 256, 0, stream>>>(cv_wd, glu_cm, conv_o);
    bn_stats_kernel<<<D_, 256, 0, stream>>>(conv_o, bn_st);
    bn_silu_kernel<<<(B_ * D_ * T_ + 255) / 256, 256, 0, stream>>>(conv_o, bn_st, bn_g, bn_b, bn_act);
    gemm_kernel<2, 3><<<g512, 256, 0, stream>>>(bn_act, cv_w2, cv_b2, x_res, x_res, nullptr, NTOK, D_, D_);

    // ---- FF module 2 ----
    ln_kernel<0><<<NTOK, 256, 0, stream>>>(x_res, ff2_g, ff2_b, lnA, nullptr);
    gemm_kernel<0, 1><<<gDFF, 256, 0, stream>>>(lnA, ff2_w1, ff2_b1, nullptr, nullptr, hbf, NTOK, DFF_, D_);
    gemm_kernel<0, 2><<<g512, 256, 0, stream>>>(hbf, ff2_w2, ff2_b2, x_res, x_res, nullptr, NTOK, D_, DFF_);

    // ---- final LayerNorm -> d_out (f32) ----
    ln_kernel<2><<<NTOK, 256, 0, stream>>>(x_res, fin_g, fin_b, nullptr, (float*)d_out);
}
